// ESIM_1683627180325
// MI455X (gfx1250) — compile-verified, matrix operand order checked by reading, not run on hardware
//
#include <hip/hip_runtime.h>
#include <math.h>

typedef __attribute__((ext_vector_type(16))) _Float16 v16h;
typedef __attribute__((ext_vector_type(8)))  _Float16 v8h;
typedef __attribute__((ext_vector_type(8)))  float    v8f;
typedef _Float16 hf;

#define WAVE 32

__device__ __forceinline__ float sigf(float x){ return 1.0f/(1.0f+expf(-x)); }

// ---------------------------------------------------------------------------
// Generic WMMA GEMM:  C[M,N] = A[M,Kp] * W[N,Kp]^T + bias ; optional relu,
// optional f32 and/or f16 outputs, optional batching (grid.y).
// A rows must be padded to Mtiles*16; W rows to Ntiles*16; Kp multiple of 32,
// padding regions zero-filled by producers.
// ---------------------------------------------------------------------------
__global__ void __launch_bounds__(256)
k_gemm(const hf* __restrict__ A, int lda, long sA,
       const hf* __restrict__ W, int ldw, long sW,
       const float* __restrict__ bias,
       float* __restrict__ C, int ldc, long sC,
       hf* __restrict__ C16, int ldc16, long sC16,
       int Mtiles, int Ntiles, int Kp, int Mreal, int Nreal, int relu)
{
  int wave = threadIdx.x >> 5;
  int tile = blockIdx.x * 8 + wave;
  if (tile >= Mtiles * Ntiles) return;
  int tm = tile / Ntiles, tn = tile % Ntiles;
  int lane = threadIdx.x & 31;
  int r = lane & 15, hi = lane >> 4;
  long b = blockIdx.y;
  const hf* Ab = A + b * sA + (long)(tm * 16 + r) * lda;
  const hf* Wb = W + b * sW + (long)(tn * 16 + r) * ldw;
  v8f acc = {};
  for (int k0 = 0; k0 < Kp; k0 += 32) {
    __builtin_prefetch(Wb + k0 + 64, 0, 1);
    v16h av, bv;
    ((v8h*)&av)[0] = *(const v8h*)(Ab + k0 + hi * 8);
    ((v8h*)&av)[1] = *(const v8h*)(Ab + k0 + 16 + hi * 8);
    ((v8h*)&bv)[0] = *(const v8h*)(Wb + k0 + hi * 8);
    ((v8h*)&bv)[1] = *(const v8h*)(Wb + k0 + 16 + hi * 8);
    acc = __builtin_amdgcn_wmma_f32_16x16x32_f16(false, av, false, bv,
                                                 (short)0, acc, false, false);
  }
  int n = tn * 16 + r;
  float bvs = (bias && n < Nreal) ? bias[n] : 0.0f;
  #pragma unroll
  for (int i = 0; i < 8; ++i) {
    int m = tm * 16 + hi * 8 + i;
    float v = acc[i] + bvs;
    if (relu) v = fmaxf(v, 0.0f);
    bool ok = (m < Mreal) && (n < Nreal);
    if (C && ok) C[b * sC + (long)m * ldc + n] = v;
    if (C16)     C16[b * sC16 + (long)m * ldc16 + n] = ok ? (hf)v : (hf)0.0f;
  }
}

// ---------------------------------------------------------------------------
// Persistent LSTM recurrence. grid.x = 2 (0=fwd,1=bwd). 512 threads.
// h state: f16 in LDS (padded 112 x Khp). c state: registers.
// Per step: WMMA  h(100xH) * Whh_packed(NG x Khp)^T -> gbuf (global, L2-hot),
// then elementwise gates, h written back to LDS + global (f16 and opt. f32).
// ---------------------------------------------------------------------------
#define MPAD 112
#define CMAX 20
__global__ void __launch_bounds__(512)
k_lstm_rec(const float* __restrict__ xwF, const float* __restrict__ xwB,
           const hf* __restrict__ WhhF, const hf* __restrict__ WhhB,
           float* __restrict__ gbuf,
           hf* __restrict__ h16, int ldh, int padCols,
           float* __restrict__ h32, int ldh32,
           int T_, int B_, int H_, int NG, int GS, int Khp)
{
  __shared__ __align__(16) hf hA[MPAD * 128];
  int tid = threadIdx.x;
  int dir = blockIdx.x;
  const float* xw = dir ? xwB : xwF;
  const hf* Whh = dir ? WhhB : WhhF;
  float* gb = gbuf + (long)dir * MPAD * 512;
  for (int i = tid; i < MPAD * 128; i += 512) hA[i] = (hf)0.0f;
  float creg[CMAX];
  #pragma unroll
  for (int s = 0; s < CMAX; ++s) creg[s] = 0.0f;
  int wave = tid >> 5, lane = tid & 31, r = lane & 15, hi = lane >> 4;
  int tilesN = NG >> 4, tilesM = (B_ + 15) >> 4, tiles = tilesM * tilesN;
  int tot = B_ * H_;
  for (int step = 0; step < T_; ++step) {
    int t = dir ? (T_ - 1 - step) : step;
    __syncthreads();                       // hA ready / gbuf consumed
    for (int tile = wave; tile < tiles; tile += 16) {
      int tm = tile / tilesN, tn = tile % tilesN;
      const hf* Ar = hA + (tm * 16 + r) * Khp;
      const hf* Wr = Whh + (long)(tn * 16 + r) * Khp;
      v8f acc = {};
      for (int k0 = 0; k0 < Khp; k0 += 32) {
        v16h av, bv;
        ((v8h*)&av)[0] = *(const v8h*)(Ar + k0 + hi * 8);
        ((v8h*)&av)[1] = *(const v8h*)(Ar + k0 + 16 + hi * 8);
        ((v8h*)&bv)[0] = *(const v8h*)(Wr + k0 + hi * 8);
        ((v8h*)&bv)[1] = *(const v8h*)(Wr + k0 + 16 + hi * 8);
        acc = __builtin_amdgcn_wmma_f32_16x16x32_f16(false, av, false, bv,
                                                     (short)0, acc, false, false);
      }
      int n = tn * 16 + r;
      #pragma unroll
      for (int i = 0; i < 8; ++i)
        gb[(long)(tm * 16 + hi * 8 + i) * NG + n] = acc[i];
    }
    __syncthreads();                       // gbuf visible to whole block
    const float* xwt = xw + ((long)t * B_) * NG;
    for (int s = 0, e = tid; e < tot; ++s, e += 512) {
      int b = e / H_, j = e % H_;
      const float* g0 = gb + (long)b * NG;
      const float* x0 = xwt + (long)b * NG;
      float gi = x0[j]          + g0[j];
      float gf = x0[GS + j]     + g0[GS + j];
      float gg = x0[2 * GS + j] + g0[2 * GS + j];
      float go = x0[3 * GS + j] + g0[3 * GS + j];
      float c = sigf(gf) * creg[s] + sigf(gi) * tanhf(gg);
      float h = sigf(go) * tanhf(c);
      creg[s] = c;
      hA[b * Khp + j] = (hf)h;
      long o = (long)t * B_ + b;
      h16[o * ldh + dir * H_ + j] = (hf)h;
      if (h32) h32[o * ldh32 + dir * H_ + j] = h;
    }
    if (dir == 0 && padCols > 0)
      for (int p = tid; p < B_ * padCols; p += 512) {
        int b = p / padCols, j = 2 * H_ + p % padCols;
        h16[((long)t * B_ + b) * ldh + j] = (hf)0.0f;
      }
  }
}

// --------------------------- small helper kernels ---------------------------
__global__ void k_pack_w(const float* __restrict__ src, hf* __restrict__ dst,
                         int H, int Din, int GS, int Kp, int NG)
{
  long n = (long)NG * Kp;
  for (long i = blockIdx.x * (long)blockDim.x + threadIdx.x; i < n;
       i += (long)gridDim.x * blockDim.x) {
    int row = (int)(i / Kp), k = (int)(i % Kp);
    int g = row / GS, j = row % GS;
    float v = (j < H && k < Din) ? src[((long)(g * H + j)) * Din + k] : 0.0f;
    dst[i] = (hf)v;
  }
}
__global__ void k_pack_b(const float* bih, const float* bhh, float* dst,
                         int H, int GS, int NG)
{
  int i = blockIdx.x * blockDim.x + threadIdx.x;
  if (i >= NG) return;
  int g = i / GS, j = i % GS;
  dst[i] = (j < H) ? bih[g * H + j] + bhh[g * H + j] : 0.0f;
}
__global__ void k_pack_fc(const float* src, hf* dst, int Nr, int Np, int K)
{
  long n = (long)Np * K;
  for (long i = blockIdx.x * (long)blockDim.x + threadIdx.x; i < n;
       i += (long)gridDim.x * blockDim.x) {
    int r = (int)(i / K), k = (int)(i % K);
    dst[i] = (r < Nr) ? (hf)src[(long)r * K + k] : (hf)0.0f;
  }
}
__global__ void k_pack_fcb(const float* src, float* dst, int Nr, int Np)
{
  int i = blockIdx.x * blockDim.x + threadIdx.x;
  if (i < Np) dst[i] = (i < Nr) ? src[i] : 0.0f;
}
__global__ void k_embed(const float* __restrict__ emb, const int* __restrict__ x,
                        hf* __restrict__ dst)
{
  long n = 6400L * 320;
  for (long i = blockIdx.x * (long)blockDim.x + threadIdx.x; i < n;
       i += (long)gridDim.x * blockDim.x) {
    long row = i / 320; int e = (int)(i % 320);
    dst[i] = (e < 300) ? (hf)emb[(long)x[row] * 300 + e] : (hf)0.0f;
  }
}
__global__ void k_zero_f32(float* p, long n)
{
  for (long i = blockIdx.x * (long)blockDim.x + threadIdx.x; i < n;
       i += (long)gridDim.x * blockDim.x) p[i] = 0.0f;
}
__global__ void k_sum(const float* __restrict__ x, long n, float* acc)
{
  __shared__ float sm[256];
  float s = 0;
  for (long i = blockIdx.x * (long)blockDim.x + threadIdx.x; i < n;
       i += (long)gridDim.x * blockDim.x) s += x[i];
  sm[threadIdx.x] = s; __syncthreads();
  for (int w = 128; w > 0; w >>= 1) {
    if ((int)threadIdx.x < w) sm[threadIdx.x] += sm[threadIdx.x + w];
    __syncthreads();
  }
  if (threadIdx.x == 0) atomicAdd(acc, sm[0]);
}
__global__ void k_scale(float* __restrict__ h32, hf* __restrict__ h16,
                        const float* sum)
{
  long n = 6400L * 224;
  float scale = fabsf(1280000.0f / *sum);   // |1/mean|
  for (long i = blockIdx.x * (long)blockDim.x + threadIdx.x; i < n;
       i += (long)gridDim.x * blockDim.x) {
    long row = i / 224; int d = (int)(i % 224);
    if (d < 200) {
      float v = h32[row * 200 + d] * scale;
      h32[row * 200 + d] = v;
      h16[i] = (hf)v;
    } else h16[i] = (hf)0.0f;
  }
}
__global__ void k_trans(const hf* __restrict__ src, hf* __restrict__ dst)
{
  long n = 100L * 224 * 64;
  for (long i = blockIdx.x * (long)blockDim.x + threadIdx.x; i < n;
       i += (long)gridDim.x * blockDim.x) {
    long b = i / (224 * 64); long rem = i % (224 * 64);
    int d = (int)(rem / 64), t = (int)(rem % 64);
    dst[i] = src[((long)t * 100 + b) * 224 + d];
  }
}
__global__ void k_softmax(const float* __restrict__ S, hf* __restrict__ P,
                          int rs, int cs)
{
  int row = blockIdx.x * 4 + ((int)threadIdx.x >> 5);
  if (row >= 6400) return;
  int lane = threadIdx.x & 31;
  int b = row >> 6, x = row & 63;
  const float* base = S + (long)b * 4096 + (long)x * rs;
  float v0 = base[lane * cs], v1 = base[(lane + 32) * cs];
  float m = fmaxf(v0, v1);
  for (int o = 16; o > 0; o >>= 1) m = fmaxf(m, __shfl_xor(m, o, 32));
  float e0 = expf(v0 - m), e1 = expf(v1 - m);
  float s = e0 + e1;
  for (int o = 16; o > 0; o >>= 1) s += __shfl_xor(s, o, 32);
  float inv = 1.0f / s;
  P[(long)row * 64 + lane]      = (hf)(e0 * inv);
  P[(long)row * 64 + lane + 32] = (hf)(e1 * inv);
}
__global__ void k_build_m(const float* __restrict__ h, const float* __restrict__ a,
                          hf* __restrict__ m)
{
  long n = 256L * 100 * 224;
  for (long i = blockIdx.x * (long)blockDim.x + threadIdx.x; i < n;
       i += (long)gridDim.x * blockDim.x) {
    long row = i / 224; int d = (int)(i % 224);
    int tp = (int)(row / 100), b = (int)(row % 100);
    if (d >= 200) { m[i] = (hf)0.0f; continue; }
    int seg = tp >> 6, tt = tp & 63;
    float hv = h[((long)tt * 100 + b) * 200 + d];
    float av = a[((long)b * 64 + tt) * 200 + d];
    float r = (seg == 0) ? hv : (seg == 1) ? av : (seg == 2) ? hv - av : hv * av;
    m[i] = (hf)r;
  }
}
__global__ void k_batchmean(const float* __restrict__ v, float* c1)
{
  __shared__ float sm[256];
  int b = blockIdx.x;
  float s = 0;
  for (int i = threadIdx.x; i < 256 * 128; i += 256) {
    int t = i >> 7, d = i & 127;
    s += v[((long)t * 100 + b) * 128 + d];
  }
  sm[threadIdx.x] = s; __syncthreads();
  for (int w = 128; w > 0; w >>= 1) {
    if ((int)threadIdx.x < w) sm[threadIdx.x] += sm[threadIdx.x + w];
    __syncthreads();
  }
  if (threadIdx.x == 0) c1[b] = fabsf(32768.0f / sm[0]);
}
__global__ void k_pool(const float* __restrict__ v1, const float* __restrict__ v2,
                       const float* __restrict__ c1, hf* __restrict__ vp)
{
  int idx = blockIdx.x * blockDim.x + threadIdx.x;
  if (idx >= 112 * 512) return;
  int b = idx >> 9, col = idx & 511;
  if (b >= 100) { vp[idx] = (hf)0.0f; return; }
  int seg = col >> 7, d = col & 127;
  const float* V = (seg == 0 || seg == 2) ? v1 : v2;
  float c = c1[b], r;
  if (seg < 2) {
    float s = 0;
    for (int t = 0; t < 256; ++t) s += V[((long)t * 100 + b) * 128 + d];
    r = s * c * (1.0f / 128.0f);
  } else {
    float mx = -3.4e38f;
    for (int t = 0; t < 256; ++t) mx = fmaxf(mx, V[((long)t * 100 + b) * 128 + d] * c);
    r = mx;
  }
  vp[idx] = (hf)r;
}
__global__ void k_final(const float* __restrict__ fcout, float* __restrict__ out)
{
  int b = blockIdx.x * blockDim.x + threadIdx.x;
  if (b >= 100) return;
  float x = fcout[b * 16], y = fcout[b * 16 + 1];
  float inv = rsqrtf(x * x + y * y);
  out[b * 2] = x * inv; out[b * 2 + 1] = y * inv;
}

// ----------------------------- host helpers --------------------------------
static inline unsigned gblocks(long n, int bs)
{
  long b = (n + bs - 1) / bs;
  if (b > 16384) b = 16384;
  if (b < 1) b = 1;
  return (unsigned)b;
}
static void launch_gemm(hipStream_t st, const hf* A, int lda, long sA,
                        const hf* W, int ldw, long sW, const float* bias,
                        float* C, int ldc, long sC, hf* C16, int ldc16, long sC16,
                        int Mreal, int Nfull, int Kp, int Nreal, int relu, int batch)
{
  int Mt = (Mreal + 15) / 16, Nt = Nfull / 16;
  long tiles = (long)Mt * Nt;
  dim3 grid((unsigned)((tiles + 7) / 8), (unsigned)batch);
  k_gemm<<<grid, 256, 0, st>>>(A, lda, sA, W, ldw, sW, bias, C, ldc, sC,
                               C16, ldc16, sC16, Mt, Nt, Kp, Mreal, Nreal, relu);
}

extern "C" void kernel_launch(void* const* d_in, const int* in_sizes, int n_in,
                              void* d_out, int out_size, void* d_ws, size_t ws_size,
                              hipStream_t stream)
{
  (void)in_sizes; (void)n_in; (void)out_size; (void)ws_size;
  char* ws = (char*)d_ws;
  size_t off = 0;
  auto alloc = [&](size_t bytes) { size_t o = off; off += (bytes + 255) & ~(size_t)255; return o; };

  // chains: 0=lstm1, 1=lstm2, 2=lstm3
  const int NGc[3] = {448, 448, 256}, GSc[3] = {112, 112, 64};
  const int Khpc[3] = {128, 128, 64}, Hc[3] = {100, 100, 64};
  const int Kin0[3] = {320, 320, 224}, Kin1[3] = {224, 224, 128};
  const int Din0[3] = {300, 300, 200}, Din1[3] = {200, 200, 128};
  const int inbase[3] = {1, 17, 33};

  size_t wih_o[3][4], whh_o[3][4], b_o[3][4];
  for (int ch = 0; ch < 3; ++ch)
    for (int c = 0; c < 4; ++c) {
      int kin = (c < 2) ? Kin0[ch] : Kin1[ch];
      wih_o[ch][c] = alloc((size_t)NGc[ch] * kin * 2);
      whh_o[ch][c] = alloc((size_t)NGc[ch] * Khpc[ch] * 2);
      b_o[ch][c]   = alloc((size_t)NGc[ch] * 4);
    }
  size_t fc1w_o = alloc(1024L * 512 * 2), fc2w_o = alloc(512L * 1024 * 2);
  size_t fc3w_o = alloc(16L * 512 * 2);
  size_t fcb1_o = alloc(1024 * 4), fcb2_o = alloc(512 * 4), fcb3_o = alloc(16 * 4);
  size_t xe1_o = alloc(6400L * 320 * 2), xe2_o = alloc(6400L * 320 * 2);
  size_t Xf_o = alloc(25600L * 256 * 4), Xb_o = alloc(25600L * 256 * 4);
  size_t gbuf_o = alloc(2L * MPAD * 512 * 4);
  size_t hmid_o = alloc(25600L * 128 * 2);
  size_t h1f16_o = alloc(6400L * 224 * 2), h2f16_o = alloc(6400L * 224 * 2);
  size_t h1f32_o = alloc(6400L * 200 * 4), h2f32_o = alloc(6400L * 200 * 4);
  size_t h1T_o = alloc(100L * 224 * 64 * 2), h2T_o = alloc(100L * 224 * 64 * 2);
  size_t S_o = alloc(100L * 4096 * 4);
  size_t P1_o = alloc(100L * 4096 * 2), P2_o = alloc(100L * 4096 * 2);
  size_t a1_o = alloc(100L * 64 * 200 * 4), a2_o = alloc(100L * 64 * 200 * 4);
  size_t m1_o = alloc(25600L * 224 * 2), m2_o = alloc(25600L * 224 * 2);
  size_t v1_o = alloc(25600L * 128 * 4), v2_o = alloc(25600L * 128 * 4);
  size_t c1_o = alloc(112 * 4);
  size_t vp_o = alloc(112L * 512 * 2);
  size_t fb1_o = alloc(112L * 1024 * 2), fb2_o = alloc(112L * 512 * 2);
  size_t fco_o = alloc(112L * 16 * 4);
  size_t acc_o = alloc(2 * 4);

  // ---- pack weights -------------------------------------------------------
  for (int ch = 0; ch < 3; ++ch)
    for (int c = 0; c < 4; ++c) {
      const float* Wih = (const float*)d_in[inbase[ch] + 4 * c + 0];
      const float* Whh = (const float*)d_in[inbase[ch] + 4 * c + 1];
      const float* bih = (const float*)d_in[inbase[ch] + 4 * c + 2];
      const float* bhh = (const float*)d_in[inbase[ch] + 4 * c + 3];
      int Din = (c < 2) ? Din0[ch] : Din1[ch];
      int Kp  = (c < 2) ? Kin0[ch] : Kin1[ch];
      k_pack_w<<<gblocks((long)NGc[ch] * Kp, 256), 256, 0, stream>>>(
          Wih, (hf*)(ws + wih_o[ch][c]), Hc[ch], Din, GSc[ch], Kp, NGc[ch]);
      k_pack_w<<<gblocks((long)NGc[ch] * Khpc[ch], 256), 256, 0, stream>>>(
          Whh, (hf*)(ws + whh_o[ch][c]), Hc[ch], Hc[ch], GSc[ch], Khpc[ch], NGc[ch]);
      k_pack_b<<<gblocks(NGc[ch], 256), 256, 0, stream>>>(
          bih, bhh, (float*)(ws + b_o[ch][c]), Hc[ch], GSc[ch], NGc[ch]);
    }
  k_pack_fc<<<gblocks(1024L * 512, 256), 256, 0, stream>>>((const float*)d_in[49], (hf*)(ws + fc1w_o), 1024, 1024, 512);
  k_pack_fc<<<gblocks(512L * 1024, 256), 256, 0, stream>>>((const float*)d_in[51], (hf*)(ws + fc2w_o), 512, 512, 1024);
  k_pack_fc<<<gblocks(16L * 512, 256), 256, 0, stream>>>((const float*)d_in[53], (hf*)(ws + fc3w_o), 2, 16, 512);
  k_pack_fcb<<<4, 256, 0, stream>>>((const float*)d_in[50], (float*)(ws + fcb1_o), 1024, 1024);
  k_pack_fcb<<<2, 256, 0, stream>>>((const float*)d_in[52], (float*)(ws + fcb2_o), 512, 512);
  k_pack_fcb<<<1, 256, 0, stream>>>((const float*)d_in[54], (float*)(ws + fcb3_o), 2, 16);

  // ---- embedding gather (f16, K padded 300->320) --------------------------
  k_embed<<<gblocks(6400L * 320, 256), 256, 0, stream>>>(
      (const float*)d_in[0], (const int*)d_in[55], (hf*)(ws + xe1_o));
  k_embed<<<gblocks(6400L * 320, 256), 256, 0, stream>>>(
      (const float*)d_in[0], (const int*)d_in[56], (hf*)(ws + xe2_o));

  float* Xf = (float*)(ws + Xf_o);
  float* Xb = (float*)(ws + Xb_o);
  float* gbuf = (float*)(ws + gbuf_o);
  hf* hmid = (hf*)(ws + hmid_o);

  auto run_layer = [&](int ch, int layer, const hf* Ain, int lda, int Min, int T_,
                       hf* h16, int ldh, int padCols, float* h32, int ldh32) {
    int cf = 2 * layer, cb = 2 * layer + 1;
    int Kp = (layer == 0) ? Kin0[ch] : Kin1[ch];
    launch_gemm(stream, Ain, lda, 0, (const hf*)(ws + wih_o[ch][cf]), Kp, 0,
                (const float*)(ws + b_o[ch][cf]), Xf, NGc[ch], 0,
                nullptr, 0, 0, Min, NGc[ch], Kp, NGc[ch], 0, 1);
    launch_gemm(stream, Ain, lda, 0, (const hf*)(ws + wih_o[ch][cb]), Kp, 0,
                (const float*)(ws + b_o[ch][cb]), Xb, NGc[ch], 0,
                nullptr, 0, 0, Min, NGc[ch], Kp, NGc[ch], 0, 1);
    k_lstm_rec<<<2, 512, 0, stream>>>(Xf, Xb,
        (const hf*)(ws + whh_o[ch][cf]), (const hf*)(ws + whh_o[ch][cb]),
        gbuf, h16, ldh, padCols, h32, ldh32,
        T_, 100, Hc[ch], NGc[ch], GSc[ch], Khpc[ch]);
  };

  // ---- BiLSTM 1 & 2 -------------------------------------------------------
  run_layer(0, 0, (const hf*)(ws + xe1_o), 320, 6400, 64, hmid, 224, 24, nullptr, 0);
  run_layer(0, 1, hmid, 224, 6400, 64, (hf*)(ws + h1f16_o), 224, 24, (float*)(ws + h1f32_o), 200);
  run_layer(1, 0, (const hf*)(ws + xe2_o), 320, 6400, 64, hmid, 224, 24, nullptr, 0);
  run_layer(1, 1, hmid, 224, 6400, 64, (hf*)(ws + h2f16_o), 224, 24, (float*)(ws + h2f32_o), 200);

  // ---- normalize by |1/mean| ---------------------------------------------
  k_zero_f32<<<1, 64, 0, stream>>>((float*)(ws + acc_o), 2);
  k_sum<<<256, 256, 0, stream>>>((const float*)(ws + h1f32_o), 6400L * 200, (float*)(ws + acc_o));
  k_sum<<<256, 256, 0, stream>>>((const float*)(ws + h2f32_o), 6400L * 200, (float*)(ws + acc_o) + 1);
  k_scale<<<gblocks(6400L * 224, 256), 256, 0, stream>>>(
      (float*)(ws + h1f32_o), (hf*)(ws + h1f16_o), (const float*)(ws + acc_o));
  k_scale<<<gblocks(6400L * 224, 256), 256, 0, stream>>>(
      (float*)(ws + h2f32_o), (hf*)(ws + h2f16_o), (const float*)(ws + acc_o) + 1);

  // ---- attention ----------------------------------------------------------
  k_trans<<<gblocks(100L * 224 * 64, 256), 256, 0, stream>>>((const hf*)(ws + h1f16_o), (hf*)(ws + h1T_o));
  k_trans<<<gblocks(100L * 224 * 64, 256), 256, 0, stream>>>((const hf*)(ws + h2f16_o), (hf*)(ws + h2T_o));
  // S[b] = h1n[b] (64x224) * h2n[b]^T
  launch_gemm(stream, (const hf*)(ws + h1f16_o), 22400, 224,
              (const hf*)(ws + h2f16_o), 22400, 224, nullptr,
              (float*)(ws + S_o), 64, 4096, nullptr, 0, 0,
              64, 64, 224, 64, 0, 100);
  k_softmax<<<1600, 128, 0, stream>>>((const float*)(ws + S_o), (hf*)(ws + P1_o), 64, 1);
  k_softmax<<<1600, 128, 0, stream>>>((const float*)(ws + S_o), (hf*)(ws + P2_o), 1, 64);
  // a1[b] = P1[b] (64x64) * h2n[b] ;  a2[b] = P2[b] * h1n[b]
  launch_gemm(stream, (const hf*)(ws + P1_o), 64, 4096,
              (const hf*)(ws + h2T_o), 64, 14336, nullptr,
              (float*)(ws + a1_o), 200, 12800, nullptr, 0, 0,
              64, 224, 64, 200, 0, 100);
  launch_gemm(stream, (const hf*)(ws + P2_o), 64, 4096,
              (const hf*)(ws + h1T_o), 64, 14336, nullptr,
              (float*)(ws + a2_o), 200, 12800, nullptr, 0, 0,
              64, 224, 64, 200, 0, 100);
  k_build_m<<<gblocks(256L * 100 * 224, 256), 256, 0, stream>>>(
      (const float*)(ws + h1f32_o), (const float*)(ws + a1_o), (hf*)(ws + m1_o));
  k_build_m<<<gblocks(256L * 100 * 224, 256), 256, 0, stream>>>(
      (const float*)(ws + h2f32_o), (const float*)(ws + a2_o), (hf*)(ws + m2_o));

  // ---- BiLSTM 3 on m1 -> v1, m2 -> v2 ------------------------------------
  run_layer(2, 0, (const hf*)(ws + m1_o), 224, 25600, 256, hmid, 128, 0, nullptr, 0);
  run_layer(2, 1, hmid, 128, 25600, 256, hmid, 128, 0, (float*)(ws + v1_o), 128);
  run_layer(2, 0, (const hf*)(ws + m2_o), 224, 25600, 256, hmid, 128, 0, nullptr, 0);
  run_layer(2, 1, hmid, 128, 25600, 256, hmid, 128, 0, (float*)(ws + v2_o), 128);

  // ---- pooling + FC head --------------------------------------------------
  k_batchmean<<<100, 256, 0, stream>>>((const float*)(ws + v1_o), (float*)(ws + c1_o));
  k_pool<<<(112 * 512 + 255) / 256, 256, 0, stream>>>(
      (const float*)(ws + v1_o), (const float*)(ws + v2_o),
      (const float*)(ws + c1_o), (hf*)(ws + vp_o));
  launch_gemm(stream, (const hf*)(ws + vp_o), 512, 0,
              (const hf*)(ws + fc1w_o), 512, 0, (const float*)(ws + fcb1_o),
              nullptr, 0, 0, (hf*)(ws + fb1_o), 1024, 0,
              100, 1024, 512, 1024, 1, 1);
  launch_gemm(stream, (const hf*)(ws + fb1_o), 1024, 0,
              (const hf*)(ws + fc2w_o), 1024, 0, (const float*)(ws + fcb2_o),
              nullptr, 0, 0, (hf*)(ws + fb2_o), 512, 0,
              100, 512, 1024, 512, 1, 1);
  launch_gemm(stream, (const hf*)(ws + fb2_o), 512, 0,
              (const hf*)(ws + fc3w_o), 512, 0, (const float*)(ws + fcb3_o),
              (float*)(ws + fco_o), 16, 0, nullptr, 0, 0,
              100, 16, 512, 2, 0, 1);
  k_final<<<1, 128, 0, stream>>>((const float*)(ws + fco_o), (float*)d_out);
}
